// MultiHeadAttention_72189810312078
// MI455X (gfx1250) — matrix operand order repaired and checked by reading, not compiled, not run on hardware
//
#include <hip/hip_runtime.h>

// ---- CDNA5 (gfx1250, wave32) WMMA fragment types ----
typedef __attribute__((ext_vector_type(16))) __bf16 v16bf;
typedef __attribute__((ext_vector_type(8)))  __bf16 v8bf;
typedef __attribute__((ext_vector_type(8)))  float  v8f;
typedef __attribute__((ext_vector_type(2)))  float  v2f;

#define NCH   1024      // IN_CHANNELS
#define NH    16        // NUM_HEADS
#define HD    64        // HEAD_DIM
#define QKVC  3072      // 3 * NCH
#define TOK   32        // tokens (rows) per workgroup = two 16-row WMMA tiles
#define WAVES 8         // waves per workgroup (blockDim = 256)

// Assemble a 16-element bf16 WMMA fragment from two contiguous 8-element
// (16-byte, b128-load) halves.
__device__ __forceinline__ v16bf frag16(v8bf lo, v8bf hi) {
    return __builtin_shufflevector(lo, hi, 0, 1, 2, 3, 4, 5, 6, 7,
                                   8, 9, 10, 11, 12, 13, 14, 15);
}

// One-time (per launch) fp32 -> bf16 weight conversion into workspace.
__global__ __launch_bounds__(256) void cvt_f32_bf16_kernel(
    const float* __restrict__ src, __bf16* __restrict__ dst, int n4) {
    const int i = (blockIdx.x * 256 + threadIdx.x) * 4;
    if (i < n4 * 4) {
        float4 v = *(const float4*)(src + i);
        dst[i]     = (__bf16)v.x;
        dst[i + 1] = (__bf16)v.y;
        dst[i + 2] = (__bf16)v.z;
        dst[i + 3] = (__bf16)v.w;
    }
}

// Fully fused: qkv GEMM (bf16 WMMA) -> per-token 16x16 head-attention
// (f32 WMMA + softmax) -> output projection (bf16 WMMA). 8 MB workspace
// holds bf16 weights; activations never round-trip to HBM.
__global__ __launch_bounds__(256) void mha_fused_kernel(
    const float*  __restrict__ x,
    const __bf16* __restrict__ wq_bf,   // [3072][1024] bf16
    const float*  __restrict__ b_qkv,
    const __bf16* __restrict__ wo_bf,   // [1024][1024] bf16
    const float*  __restrict__ b_out,
    float*        __restrict__ out)
{
    // 192 KB + 64 KB + 8 KB = 264 KB (< 320 KB/WGP)
    __shared__ __bf16 qkvl[TOK][QKVC];  // qkv tile, bf16
    __shared__ __bf16 xh[TOK][NCH];     // stage 0/1: x tile; stage 2/3: h tile
    __shared__ float  ls[WAVES][16][16];// per-wave softmax'd scores

    const int  tid  = threadIdx.x;
    const int  wave = tid >> 5;
    const int  lane = tid & 31;
    const int  half = lane >> 4;        // which 16-lane half of the wave
    const int  l16  = lane & 15;
    const long tok0 = (long)blockIdx.x * TOK;

    // ---------- Stage 0: stage x tile into LDS as bf16 (coalesced) ----------
    for (int i = tid; i < TOK * NCH / 4; i += 256) {
        const int e   = i * 4;
        const int row = e >> 10;        // / NCH
        const int col = e & (NCH - 1);
        float4 v = *(const float4*)(x + (tok0 + row) * NCH + col);
        xh[row][col]     = (__bf16)v.x;
        xh[row][col + 1] = (__bf16)v.y;
        xh[row][col + 2] = (__bf16)v.z;
        xh[row][col + 3] = (__bf16)v.w;
    }
    __syncthreads();

    // ---------------- Stage 1: qkv = x @ w_qkv^T + b_qkv ----------------
    // A (16x32 bf16): lane = token row, elems 0..7 = K base+0..7,
    // elems 8..15 = K base+16..23, base = k0 + half*8 (two ds_load_b128).
    // B (32x16 bf16): lane = out column, same K order as A: elems 0..7 =
    // kb+0..7, elems 8..15 = kb+16..23, kb = k0 + half*8 (two global_load_b128).
    for (int t = wave; t < QKVC / 16; t += WAVES) {
        const int     col0 = t * 16;
        const __bf16* wrow = wq_bf + (long)(col0 + l16) * NCH;
        // prefetch next col-tile's weight row (global_prefetch_b8)
        __builtin_prefetch(wrow + (long)(WAVES * 16) * NCH, 0, 0);
        v8f acc0 = {0.f,0.f,0.f,0.f,0.f,0.f,0.f,0.f};
        v8f acc1 = {0.f,0.f,0.f,0.f,0.f,0.f,0.f,0.f};
        for (int k0 = 0; k0 < NCH; k0 += 32) {
            const int ka = k0 + half * 8;
            v16bf a0 = frag16(*(const v8bf*)&xh[l16][ka],
                              *(const v8bf*)&xh[l16][ka + 16]);
            v16bf a1 = frag16(*(const v8bf*)&xh[16 + l16][ka],
                              *(const v8bf*)&xh[16 + l16][ka + 16]);
            const int kb = k0 + half * 8;
            v16bf b  = frag16(*(const v8bf*)(wrow + kb),
                              *(const v8bf*)(wrow + kb + 16));
            acc0 = __builtin_amdgcn_wmma_f32_16x16x32_bf16(false, a0, false, b,
                                                           (short)0, acc0, false, false);
            acc1 = __builtin_amdgcn_wmma_f32_16x16x32_bf16(false, a1, false, b,
                                                           (short)0, acc1, false, false);
        }
        const float bias = b_qkv[col0 + l16];
        #pragma unroll
        for (int r = 0; r < 8; ++r) {   // C/D layout: VGPR r -> M = r + half*8
            qkvl[r + half * 8][col0 + l16]      = (__bf16)(acc0[r] + bias);
            qkvl[16 + r + half * 8][col0 + l16] = (__bf16)(acc1[r] + bias);
        }
    }
    __syncthreads();   // qkvl complete; xh (x tile) now dead

    // ------- Stage 2: per-token attention over heads (f32 WMMA) -------
    // Per token: Q,K,V are [16 heads x 64]; S = softmax(scale*Q K^T); O = S V.
    const float scale = 0.125f;  // HEAD_DIM^-0.5
    for (int t = wave; t < TOK; t += WAVES) {
        const __bf16* qt = &qkvl[t][0];  // [head][192] = {q[64], k[64], v[64]}
        v8f s = {0.f,0.f,0.f,0.f,0.f,0.f,0.f,0.f};
        for (int d0 = 0; d0 < HD; d0 += 4) {
            v2f av, bv;
            const int d = d0 + 2 * half;  // f32 A/B: V0 -> K = d0 + 2*half
            av.x = (float)qt[l16 * 192 + d]          * scale;   // q[h=l16][d]
            av.y = (float)qt[l16 * 192 + d + 1]      * scale;
            bv.x = (float)qt[l16 * 192 + 64 + d];               // k[g=l16][d]
            bv.y = (float)qt[l16 * 192 + 64 + d + 1];
            s = __builtin_amdgcn_wmma_f32_16x16x4_f32(false, av, false, bv,
                                                      (short)0, s, false, false);
        }
        // softmax over g (= lanes inside each 16-lane half), rows M = r+half*8
        #pragma unroll
        for (int r = 0; r < 8; ++r) {
            float m = s[r];
            m = fmaxf(m, __shfl_xor(m, 1, 16));
            m = fmaxf(m, __shfl_xor(m, 2, 16));
            m = fmaxf(m, __shfl_xor(m, 4, 16));
            m = fmaxf(m, __shfl_xor(m, 8, 16));
            float e   = __expf(s[r] - m);
            float sum = e;
            sum += __shfl_xor(sum, 1, 16);
            sum += __shfl_xor(sum, 2, 16);
            sum += __shfl_xor(sum, 4, 16);
            sum += __shfl_xor(sum, 8, 16);
            s[r] = e / sum;
        }
        #pragma unroll
        for (int r = 0; r < 8; ++r)
            ls[wave][r + half * 8][l16] = s[r];

        // O[h, d] = sum_g S[h,g] * v[g,d]   (f32 WMMA, K = g in steps of 4)
        for (int dt = 0; dt < 4; ++dt) {
            v8f o = {0.f,0.f,0.f,0.f,0.f,0.f,0.f,0.f};
            for (int g0 = 0; g0 < 16; g0 += 4) {
                v2f av, bv;
                const int g = g0 + 2 * half;
                av.x = ls[wave][l16][g];                          // S[h=l16][g]
                av.y = ls[wave][l16][g + 1];
                bv.x = (float)qt[g * 192 + 128 + dt * 16 + l16];  // v[g][d=n]
                bv.y = (float)qt[(g + 1) * 192 + 128 + dt * 16 + l16];
                o = __builtin_amdgcn_wmma_f32_16x16x4_f32(false, av, false, bv,
                                                          (short)0, o, false, false);
            }
            #pragma unroll
            for (int r = 0; r < 8; ++r)   // h[t, head*64 + d], head = r+half*8
                xh[t][(r + half * 8) * HD + dt * 16 + l16] = (__bf16)o[r];
        }
    }
    __syncthreads();   // xh now holds the attention output tile h

    // ------------- Stage 3: out = h @ w_out^T + b_out -------------
    for (int t = wave; t < NCH / 16; t += WAVES) {
        const int     col0 = t * 16;
        const __bf16* wrow = wo_bf + (long)(col0 + l16) * NCH;
        __builtin_prefetch(wrow + (long)(WAVES * 16) * NCH, 0, 0);
        v8f acc0 = {0.f,0.f,0.f,0.f,0.f,0.f,0.f,0.f};
        v8f acc1 = {0.f,0.f,0.f,0.f,0.f,0.f,0.f,0.f};
        for (int k0 = 0; k0 < NCH; k0 += 32) {
            const int ka = k0 + half * 8;
            v16bf a0 = frag16(*(const v8bf*)&xh[l16][ka],
                              *(const v8bf*)&xh[l16][ka + 16]);
            v16bf a1 = frag16(*(const v8bf*)&xh[16 + l16][ka],
                              *(const v8bf*)&xh[16 + l16][ka + 16]);
            const int kb = k0 + half * 8;
            v16bf b  = frag16(*(const v8bf*)(wrow + kb),
                              *(const v8bf*)(wrow + kb + 16));
            acc0 = __builtin_amdgcn_wmma_f32_16x16x32_bf16(false, a0, false, b,
                                                           (short)0, acc0, false, false);
            acc1 = __builtin_amdgcn_wmma_f32_16x16x32_bf16(false, a1, false, b,
                                                           (short)0, acc1, false, false);
        }
        const float bias = b_out[col0 + l16];
        #pragma unroll
        for (int r = 0; r < 8; ++r) {
            out[(tok0 + r + half * 8) * NCH + col0 + l16]      = acc0[r] + bias;
            out[(tok0 + 16 + r + half * 8) * NCH + col0 + l16] = acc1[r] + bias;
        }
    }
}

extern "C" void kernel_launch(void* const* d_in, const int* in_sizes, int n_in,
                              void* d_out, int out_size, void* d_ws, size_t ws_size,
                              hipStream_t stream) {
    (void)n_in; (void)out_size; (void)ws_size;
    const float* x     = (const float*)d_in[0];
    const float* w_qkv = (const float*)d_in[1];
    const float* b_qkv = (const float*)d_in[2];
    const float* w_out = (const float*)d_in[3];
    const float* b_out = (const float*)d_in[4];
    float* outp = (float*)d_out;

    // Workspace: bf16 copies of the two weight matrices (8 MB total).
    __bf16* wq_bf = (__bf16*)d_ws;                        // 3072*1024 bf16
    __bf16* wo_bf = wq_bf + (size_t)QKVC * NCH;           // 1024*1024 bf16

    const int nq4 = QKVC * NCH / 4;                       // 786432 quads
    const int no4 = NCH * NCH / 4;                        // 262144 quads
    cvt_f32_bf16_kernel<<<(nq4 + 255) / 256, 256, 0, stream>>>(w_qkv, wq_bf, nq4);
    cvt_f32_bf16_kernel<<<(no4 + 255) / 256, 256, 0, stream>>>(w_out, wo_bf, no4);

    const int n = in_sizes[0] / NCH;                      // 65536 tokens
    mha_fused_kernel<<<n / TOK, 256, 0, stream>>>(
        x, wq_bf, b_qkv, wo_bf, b_out, outp);
}
